// FusedMoE_68771016343990
// MI455X (gfx1250) — matrix operand order repaired and checked by reading, not compiled, not run on hardware
//
#include <hip/hip_runtime.h>
#include <hip/hip_bf16.h>

#define E_ 8
#define K_ 2
#define H_ 2048
#define I_ 1408
#define T_ 4096
#define HCHUNK 256
#define MTILE 16
#define ITILES_PER_WAVE 11   // 8 waves * 11 tiles * 16 = 1408 = I_
#define MAXTILES (T_ / MTILE)

typedef __attribute__((ext_vector_type(16))) __bf16 v16bf;
typedef __attribute__((ext_vector_type(8)))  __bf16 v8bf;
typedef __attribute__((ext_vector_type(8)))  float  v8f;

// ---------------- helpers ----------------

// Load 16 contiguous fp32 and convert to a 16-element bf16 fragment.
// This is the per-lane payload of the B matrix (32x16 bf16): lane L holds
// column N = L%16, K = (L/16)*16 + j for j = 0..15 (contiguous along K).
__device__ __forceinline__ v16bf load_b_frag(const float* __restrict__ p) {
    const float4* q = (const float4*)p;
    float4 f0 = q[0], f1 = q[1], f2 = q[2], f3 = q[3];
    v16bf b;
    b[0]  = (__bf16)f0.x; b[1]  = (__bf16)f0.y; b[2]  = (__bf16)f0.z; b[3]  = (__bf16)f0.w;
    b[4]  = (__bf16)f1.x; b[5]  = (__bf16)f1.y; b[6]  = (__bf16)f1.z; b[7]  = (__bf16)f1.w;
    b[8]  = (__bf16)f2.x; b[9]  = (__bf16)f2.y; b[10] = (__bf16)f2.z; b[11] = (__bf16)f2.w;
    b[12] = (__bf16)f3.x; b[13] = (__bf16)f3.y; b[14] = (__bf16)f3.z; b[15] = (__bf16)f3.w;
    return b;
}

// Build A fragment (16x32 bf16) from an LDS row. Lane L has
// row M = L%16; elements j = 0..15 are K = (L/16)*16 + j (contiguous, matching B).
// Two 16-byte LDS reads per lane.
__device__ __forceinline__ v16bf load_a_frag(const __bf16* __restrict__ row, int kbase, int half) {
    int b0 = kbase + half * 16;
    v8bf lo = *(const v8bf*)(row + b0);
    v8bf hi = *(const v8bf*)(row + b0 + 8);
    v16bf a;
#pragma unroll
    for (int j = 0; j < 8; j++) { a[j] = lo[j]; a[j + 8] = hi[j]; }
    return a;
}

// ---------------- kernels ----------------

__global__ void zero_kernel(float* __restrict__ out, int n) {
    int i = blockIdx.x * blockDim.x + threadIdx.x;
    if (i < n) out[i] = 0.0f;
}

// One thread per token: top-2 of 8 logits; renormalized softmax over the pair.
__global__ void router_kernel(const float* __restrict__ logits,
                              int* __restrict__ tids, float* __restrict__ tws) {
    int t = blockIdx.x * blockDim.x + threadIdx.x;
    if (t >= T_) return;
    float l[E_];
#pragma unroll
    for (int e = 0; e < E_; e++) l[e] = logits[t * E_ + e];
    int i0 = 0; float m0 = l[0];
#pragma unroll
    for (int e = 1; e < E_; e++) if (l[e] > m0) { m0 = l[e]; i0 = e; }
    int i1 = -1; float m1 = -3.0e38f;
#pragma unroll
    for (int e = 0; e < E_; e++) if (e != i0 && l[e] > m1) { m1 = l[e]; i1 = e; }
    // p0/(p0+p1) with shared softmax denominator cancelled
    float w0 = 1.0f / (1.0f + __expf(m1 - m0));
    tids[t * 2 + 0] = i0;  tids[t * 2 + 1] = i1;
    tws [t * 2 + 0] = w0;  tws [t * 2 + 1] = 1.0f - w0;
}

// Deterministic bucketing: 1 block of 8 waves; wave e compacts, in token order,
// all tokens that routed to expert e (ordered ballot scan — wave32).
__global__ void bucket_kernel(const int* __restrict__ tids, const float* __restrict__ tws,
                              int* __restrict__ lists, float* __restrict__ lweights,
                              int* __restrict__ counts) {
    int wave = threadIdx.x >> 5;
    int lane = threadIdx.x & 31;
    if (wave >= E_) return;
    int e = wave;
    int cnt = 0;
    for (int t0 = 0; t0 < T_; t0 += 32) {
        int t = t0 + lane;
        int a = tids[t * 2 + 0];
        int b = tids[t * 2 + 1];
        bool pick = (a == e) || (b == e);
        unsigned mask = (unsigned)__ballot(pick);
        int pre = __popc(mask & ((1u << lane) - 1u));
        if (pick) {
            lists   [e * T_ + cnt + pre] = t;
            lweights[e * T_ + cnt + pre] = (a == e) ? tws[t * 2 + 0] : tws[t * 2 + 1];
        }
        cnt += __popc(mask);
    }
    if (lane == 0) counts[e] = cnt;
}

// Fused expert kernel: one workgroup = (expert e, tile of 16 tokens of e's list).
// Phase A: act = silu(x*W13_gate) * (x*W13_up) into LDS (bf16).
// Phase B: out += w_route * (act * W2), via f32 atomics.
__launch_bounds__(256, 1)
__global__ void moe_kernel(const float* __restrict__ x,
                           const float* __restrict__ w13,
                           const float* __restrict__ w2,
                           const int*   __restrict__ lists,
                           const float* __restrict__ lweights,
                           const int*   __restrict__ counts,
                           float* __restrict__ out) {
    const int e    = blockIdx.y;
    const int tile = blockIdx.x;
    const int cnt  = counts[e];
    if (tile * MTILE >= cnt) return;

    __shared__ __align__(16) __bf16 xs[MTILE][HCHUNK + 8];   //  8.25 KB
    __shared__ __align__(16) __bf16 acts[MTILE][I_ + 8];     // 44.25 KB
    __shared__ int   toks[MTILE];
    __shared__ float wts[MTILE];

    const int tid  = threadIdx.x;
    const int wave = tid >> 5;
    const int lane = tid & 31;
    const int half = lane >> 4;   // K-half selector for A/B fragments
    const int ln16 = lane & 15;   // row (A) / col (B,C) index

    if (tid < MTILE) {
        int idx = tile * MTILE + tid;
        if (idx < cnt) { toks[tid] = lists[e * T_ + idx]; wts[tid] = lweights[e * T_ + idx]; }
        else           { toks[tid] = -1;                  wts[tid] = 0.0f; }
    }
    __syncthreads();

    const float* w13e = w13 + (size_t)e * (2 * I_) * H_;
    const float* w2e  = w2  + (size_t)e * H_ * I_;

    // Persistent accumulators: 11 i-tiles per wave, gate + up.
    v8f gacc[ITILES_PER_WAVE], uacc[ITILES_PER_WAVE];
    const v8f vzero = {0.f, 0.f, 0.f, 0.f, 0.f, 0.f, 0.f, 0.f};
#pragma unroll
    for (int t = 0; t < ITILES_PER_WAVE; t++) { gacc[t] = vzero; uacc[t] = vzero; }

    // ---------------- Phase A: h = x * W13 (gate|up), K over H in chunks ----
    for (int h0 = 0; h0 < H_; h0 += HCHUNK) {
        __syncthreads();
        {   // stage 16 x HCHUNK chunk of x as bf16 into LDS (coalesced)
            int row = tid >> 4;          // 0..15 token row
            int seg = (tid & 15) * 16;   // 16 cols per thread
            int tok = toks[row];
            if (tok >= 0) {
                const float* px = x + (size_t)tok * H_ + h0 + seg;
#pragma unroll
                for (int j = 0; j < 16; j++) xs[row][seg + j] = (__bf16)px[j];
            } else {
#pragma unroll
                for (int j = 0; j < 16; j++) xs[row][seg + j] = (__bf16)0.0f;
            }
        }
        __syncthreads();

        // Prebuild the 8 A fragments of this chunk (reused by all 11 i-tiles).
        v16bf afrag[HCHUNK / 32];
#pragma unroll
        for (int s = 0; s < HCHUNK / 32; s++)
            afrag[s] = load_a_frag(&xs[ln16][0], s * 32, half);

#pragma unroll
        for (int t = 0; t < ITILES_PER_WAVE; t++) {
            const int i0 = (wave * ITILES_PER_WAVE + t) * 16;
            const float* gp = w13e + (size_t)(i0 + ln16) * H_ + h0 + half * 16;
            const float* up = gp + (size_t)I_ * H_;
            __builtin_prefetch(gp + HCHUNK, 0, 1);
#pragma unroll
            for (int s = 0; s < HCHUNK / 32; s++) {
                v16bf bg = load_b_frag(gp + s * 32);
                v16bf bu = load_b_frag(up + s * 32);
                gacc[t] = __builtin_amdgcn_wmma_f32_16x16x32_bf16(
                    false, afrag[s], false, bg, (short)0, gacc[t], false, false);
                uacc[t] = __builtin_amdgcn_wmma_f32_16x16x32_bf16(
                    false, afrag[s], false, bu, (short)0, uacc[t], false, false);
            }
        }
    }

    // SiLU(gate) * up -> acts (bf16 LDS). C layout: VGPR v, lane L -> M = v + 8*(L/16), N = L%16.
#pragma unroll
    for (int t = 0; t < ITILES_PER_WAVE; t++) {
        const int i0 = (wave * ITILES_PER_WAVE + t) * 16;
#pragma unroll
        for (int v = 0; v < 8; v++) {
            float g = gacc[t][v], u = uacc[t][v];
            float a = (g / (1.0f + __expf(-g))) * u;
            acts[v + half * 8][i0 + ln16] = (__bf16)a;
        }
    }
    __syncthreads();

    // ---------------- Phase B: out += w * (act * W2), K over I ----------------
    for (int nt = 0; nt < 16; nt++) {
        const int n0 = (wave * 16 + nt) * 16;   // 8 waves * 16 tiles * 16 = 2048 = H
        v8f acc = vzero;
        const float* bp = w2e + (size_t)(n0 + ln16) * I_ + half * 16;
#pragma unroll 4
        for (int s = 0; s < I_ / 32; s++) {     // 44 K-steps
            v16bf a = load_a_frag(&acts[ln16][0], s * 32, half);
            v16bf b = load_b_frag(bp + s * 32);
            acc = __builtin_amdgcn_wmma_f32_16x16x32_bf16(
                false, a, false, b, (short)0, acc, false, false);
        }
#pragma unroll
        for (int v = 0; v < 8; v++) {
            int M = v + half * 8;
            int tok = toks[M];
            if (tok >= 0)
                atomicAdd(out + (size_t)tok * H_ + n0 + ln16, acc[v] * wts[M]);
        }
    }
}

// ---------------- launcher ----------------

extern "C" void kernel_launch(void* const* d_in, const int* in_sizes, int n_in,
                              void* d_out, int out_size, void* d_ws, size_t ws_size,
                              hipStream_t stream) {
    const float* x      = (const float*)d_in[0];
    const float* logits = (const float*)d_in[1];
    const float* w13    = (const float*)d_in[2];
    const float* w2     = (const float*)d_in[3];
    float* out = (float*)d_out;

    char* ws = (char*)d_ws;
    int*   tids    = (int*)ws;     ws += (size_t)T_ * 2 * sizeof(int);
    float* tws     = (float*)ws;   ws += (size_t)T_ * 2 * sizeof(float);
    int*   lists   = (int*)ws;     ws += (size_t)E_ * T_ * sizeof(int);
    float* lwts    = (float*)ws;   ws += (size_t)E_ * T_ * sizeof(float);
    int*   counts  = (int*)ws;     ws += 64 * sizeof(int);

    zero_kernel<<<(T_ * H_ + 1023) / 1024, 1024, 0, stream>>>(out, T_ * H_);
    router_kernel<<<(T_ + 255) / 256, 256, 0, stream>>>(logits, tids, tws);
    bucket_kernel<<<1, 256, 0, stream>>>(tids, tws, lists, lwts, counts);

    dim3 grid(MAXTILES, E_);   // x = tile (same expert adjacent -> L2 reuse of weights)
    moe_kernel<<<grid, 256, 0, stream>>>(x, w13, w2, lists, lwts, counts, out);
}
